// FNO1dLSTM_1468878815916
// MI455X (gfx1250) — matrix operand order repaired and checked by reading, not compiled, not run on hardware
//
#include <hip/hip_runtime.h>
#include <hip/hip_bf16.h>
#include <cstddef>

typedef __attribute__((ext_vector_type(16))) _Float16 v16h;
typedef __attribute__((ext_vector_type(8)))  _Float16 v8h;
typedef __attribute__((ext_vector_type(8)))  float    v8f;

// ---------------------------------------------------------------------------
// Elementwise helpers
// ---------------------------------------------------------------------------
__global__ void cvt_f32_f16(const float* __restrict__ src, _Float16* __restrict__ dst, int n) {
    int i = blockIdx.x * 256 + threadIdx.x;
    if (i < n) dst[i] = (_Float16)src[i];
}

__global__ void bias_sum(const float* __restrict__ a, const float* __restrict__ b,
                         float* __restrict__ out, int n) {
    int i = blockIdx.x * 256 + threadIdx.x;
    if (i < n) out[i] = a[i] + b[i];
}

__global__ void zero_u32(unsigned* __restrict__ p, int n) {
    int i = blockIdx.x * 64 + threadIdx.x;
    if (i < n) p[i] = 0u;
}

// wt[o][i] = spec_wr[l][i][o][0] + lin_w[l][o][i]   (fused FNO spectral+linear weight)
__global__ void fno_combine(const float* __restrict__ spec_wr, const float* __restrict__ lin_w,
                            _Float16* __restrict__ wt, int layer) {
    const int W = 256, Mm = 16;
    int idx = blockIdx.x * 256 + threadIdx.x;     // 65536 elements
    int o = idx >> 8, i = idx & 255;
    float v = spec_wr[((size_t)(layer * W + i) * W + o) * Mm] +
              lin_w[(size_t)(layer * W + o) * W + i];
    wt[idx] = (_Float16)v;
}

// ---------------------------------------------------------------------------
// Generic WMMA GEMM:  C[M,N] = act( A[M,K](f16) * Bt[N,K](f16)^T + bias[N] )
// One wave computes a 16x64 tile (1 A tile reused over 4 B tiles).
// ACT: 0 = none, 1 = exact-erf GELU
// ---------------------------------------------------------------------------
template <int ACT>
__global__ void __launch_bounds__(256)
wmma_gemm(const _Float16* __restrict__ A, const _Float16* __restrict__ Bt,
          const float* __restrict__ bias, float* __restrict__ Cf,
          _Float16* __restrict__ Ch, int M, int N, int K) {
    const int lane = threadIdx.x & 31;
    const int wave = threadIdx.x >> 5;
    const int nG   = N >> 6;                       // groups of 64 columns
    int wt = blockIdx.x * 8 + wave;
    int total = (M >> 4) * nG;
    if (wt >= total) return;
    int mT = wt / nG;
    int n0 = (wt - mT * nG) << 6;

    // A operand: lane (l&15)=row-in-tile, (l>>4) selects K-half; two b128 loads / k-step
    const _Float16* arow = A + (size_t)(mT * 16 + (lane & 15)) * K + (lane >> 4) * 16;
    // B operand: lane (l&15)=col-in-tile, (l>>4)*16 = K-base; one 32B load / tile / k-step
    const _Float16* brow = Bt + (size_t)(n0 + (lane & 15)) * K + (lane >> 4) * 16;

    v8f acc[4] = {};
    for (int kk = 0; kk < K; kk += 32) {
        v8h alo = *(const v8h*)(arow + kk);
        v8h ahi = *(const v8h*)(arow + kk + 8);
        v16h a = __builtin_shufflevector(alo, ahi, 0,1,2,3,4,5,6,7,8,9,10,11,12,13,14,15);
#pragma unroll
        for (int nt = 0; nt < 4; ++nt) {
            v16h b = *(const v16h*)(brow + (size_t)nt * 16 * K + kk);
            acc[nt] = __builtin_amdgcn_wmma_f32_16x16x32_f16(
                false, a, false, b, (short)0, acc[nt], false, false);
        }
    }

    // Epilogue: D layout — VGPR v holds M = v + 8*(lane>>4); lane&15 = N
    const int nD = lane & 15, khD = lane >> 4;
#pragma unroll
    for (int nt = 0; nt < 4; ++nt) {
        int col = n0 + nt * 16 + nD;
        float bv = bias ? bias[col] : 0.0f;
#pragma unroll
        for (int v = 0; v < 8; ++v) {
            int row = mT * 16 + v + 8 * khD;
            float x = acc[nt][v] + bv;
            if (ACT == 1) x = 0.5f * x * (1.0f + erff(x * 0.70710678118f));
            size_t off = (size_t)row * N + col;
            if (Cf) Cf[off] = x;
            if (Ch) Ch[off] = (_Float16)x;
        }
    }
}

// ---------------------------------------------------------------------------
// Device-wide barrier: monotonically increasing counter in L2 (zeroed per
// launch by zero_u32), one arrival per workgroup, s_sleep in the spin.
// ---------------------------------------------------------------------------
__device__ __forceinline__ void device_barrier(unsigned* cnt, unsigned target) {
    __threadfence();                    // release stores to L2 (device scope)
    __syncthreads();
    if (threadIdx.x == 0) {
        atomicAdd(cnt, 1u);
        while (__hip_atomic_load(cnt, __ATOMIC_ACQUIRE, __HIP_MEMORY_SCOPE_AGENT) < target)
            __builtin_amdgcn_s_sleep(1);
    }
    __syncthreads();
}

// ---------------------------------------------------------------------------
// Multi-WGP LSTM recurrence. B=32, H=512, 4H=2048, T=384. Grid = 8 WGs x 1024.
//   wg>>2 : batch group (rows 0-15 / 16-31)  [batch dim is sync-free parallel]
//   wg&3  : 128-column h slice (and matching i/f/g/o gate column slices)
// h exchanged through a triple-buffered global f16 buffer hx[3][32][512];
// step t reads hx[t%3], writes hx[(t+1)%3]; ONE device barrier per step.
// c lives in registers (2 cells/thread, stable ownership). Gates staged via
// padded LDS between the WMMA phase and the elementwise update phase.
// ---------------------------------------------------------------------------
template <bool WRITE_SEQ>
__global__ void __launch_bounds__(1024)
lstm_recur_mw(const _Float16* __restrict__ Gx,    // [(b*T+t)][2048] f16 (x-proj + biases)
              const _Float16* __restrict__ WhhT,  // [2048][512] f16
              const float* __restrict__ h0, const float* __restrict__ c0,  // [32][512]
              _Float16* __restrict__ hx,          // [3][32][512] f16
              unsigned* __restrict__ bar,         // one counter cell
              _Float16* __restrict__ seq,         // [(b*T+t)][512] f16 or null
              float* __restrict__ hTf, float* __restrict__ cTf,
              _Float16* __restrict__ hTh, int T) {
    constexpr int H = 512, G4 = 2048, NWG = 8, HPAD = 132;
    __shared__ __attribute__((aligned(64))) float gS[4][16][HPAD];

    const int wg = blockIdx.x;
    const int bg = wg >> 2;              // batch group
    const int hg = wg & 3;               // h-column group
    const int tid = threadIdx.x;
    const int lane = tid & 31, wave = tid >> 5;
    const int gate = wave >> 3;          // 0..3 (i,f,g,o)
    const int hsub = wave & 7;           // 16-col sub-slice within the 128

    // ---- update-phase cell ownership: 16 rows x 128 cols = 2048 cells / 1024 thr
    const int r0 = tid >> 7;             // 0..7 ; second cell at row r0+8
    const int cl = tid & 127;
    const int col = hg * 128 + cl;       // global h column
    float creg[2];
    creg[0] = c0[(bg * 16 + r0) * H + col];
    creg[1] = c0[(bg * 16 + r0 + 8) * H + col];

    // ---- init hx[0] with h0 (each WG writes its own slice)
    hx[(size_t)(bg * 16 + r0) * H + col]     = (_Float16)h0[(bg * 16 + r0) * H + col];
    hx[(size_t)(bg * 16 + r0 + 8) * H + col] = (_Float16)h0[(bg * 16 + r0 + 8) * H + col];
    device_barrier(bar, NWG);

    // ---- WMMA operand setup (one 16x16 tile per wave)
    const int nL = lane & 15, khi = lane >> 4;
    const int ncol = gate * 512 + hg * 128 + hsub * 16;     // gate-column tile base
    const _Float16* brow = WhhT + (size_t)(ncol + nL) * H + khi * 16;
    const size_t arowoff = (size_t)(bg * 16 + nL) * H + khi * 16;

    for (int t = 0; t < T; ++t) {
        // prefetch this step's Gx lines so the update phase hits hot cache
        {
            size_t g0 = ((size_t)(bg * 16 + r0) * T + t) * G4 + col;
            size_t g1 = ((size_t)(bg * 16 + r0 + 8) * T + t) * G4 + col;
#pragma unroll
            for (int g = 0; g < 4; ++g) {
                __builtin_prefetch(Gx + g0 + g * 512, 0, 3);
                __builtin_prefetch(Gx + g1 + g * 512, 0, 3);
            }
        }
        const _Float16* arow = hx + (size_t)(t % 3) * 32 * H + arowoff;
        v8f acc = {};
        for (int kk = 0; kk < H; kk += 32) {
            v8h alo = *(const v8h*)(arow + kk);
            v8h ahi = *(const v8h*)(arow + kk + 8);
            v16h a = __builtin_shufflevector(alo, ahi, 0,1,2,3,4,5,6,7,8,9,10,11,12,13,14,15);
            v16h b = *(const v16h*)(brow + kk);
            acc = __builtin_amdgcn_wmma_f32_16x16x32_f16(
                false, a, false, b, (short)0, acc, false, false);
        }
        // stage gate tile to LDS (D layout: VGPR v -> row v+8*khi, lane&15 -> col)
#pragma unroll
        for (int v = 0; v < 8; ++v)
            gS[gate][v + 8 * khi][hsub * 16 + nL] = acc[v];
        __syncthreads();

        // ---- cell update: 2 cells per thread
#pragma unroll
        for (int cp = 0; cp < 2; ++cp) {
            int rr = r0 + cp * 8;
            int b  = bg * 16 + rr;
            size_t gbase = ((size_t)b * T + t) * G4 + col;
            float gi = gS[0][rr][cl] + (float)Gx[gbase];
            float gf = gS[1][rr][cl] + (float)Gx[gbase + 512];
            float gg = gS[2][rr][cl] + (float)Gx[gbase + 1024];
            float go = gS[3][rr][cl] + (float)Gx[gbase + 1536];
            float i_ = 1.0f / (1.0f + __expf(-gi));
            float f_ = 1.0f / (1.0f + __expf(-gf));
            float g_ = tanhf(gg);
            float o_ = 1.0f / (1.0f + __expf(-go));
            float cn = f_ * creg[cp] + i_ * g_;
            creg[cp] = cn;
            float hn = o_ * tanhf(cn);
            _Float16 hh = (_Float16)hn;
            hx[(size_t)(((t + 1) % 3) * 32 + b) * H + col] = hh;
            if (WRITE_SEQ) seq[((size_t)b * T + t) * H + col] = hh;
            if (t == T - 1) {
                hTf[b * H + col] = hn;
                cTf[b * H + col] = cn;
                if (hTh) hTh[b * H + col] = hh;
            }
        }
        device_barrier(bar, (unsigned)(NWG * (t + 2)));
    }
}

// ---------------------------------------------------------------------------
// Host orchestration
// ---------------------------------------------------------------------------
extern "C" void kernel_launch(void* const* d_in, const int* in_sizes, int n_in,
                              void* d_out, int out_size, void* d_ws, size_t ws_size,
                              hipStream_t stream) {
    (void)in_sizes; (void)n_in; (void)out_size; (void)ws_size;
    const int B = 32, S = 384, D = 64, W = 256, H = 512, OUT = 64;
    const int R = B * S;           // 12288 rows
    const int G4 = 4 * H;          // 2048

    const float* x       = (const float*)d_in[0];
    const float* h0      = (const float*)d_in[1];
    const float* c0      = (const float*)d_in[2];
    const float* fc0_w   = (const float*)d_in[3];
    const float* fc0_b   = (const float*)d_in[4];
    const float* spec_wr = (const float*)d_in[5];
    /* spec_wi (d_in[6]) mathematically unused: irfft(n=1) keeps only Re */
    const float* lin_w   = (const float*)d_in[7];
    const float* lin_b   = (const float*)d_in[8];
    const float* wih0    = (const float*)d_in[9];
    const float* whh0    = (const float*)d_in[10];
    const float* bih0    = (const float*)d_in[11];
    const float* bhh0    = (const float*)d_in[12];
    const float* wih1    = (const float*)d_in[13];
    const float* whh1    = (const float*)d_in[14];
    const float* bih1    = (const float*)d_in[15];
    const float* bhh1    = (const float*)d_in[16];
    const float* fcout_w = (const float*)d_in[17];
    const float* fcout_b = (const float*)d_in[18];

    // output layout: out[32,64] | h1[32,512] h2[32,512] | c1 c2
    float* out_ = (float*)d_out;
    float* h1 = out_ + B * OUT;
    float* h2 = h1 + B * H;
    float* c1 = h2 + B * H;
    float* c2 = c1 + B * H;

    // workspace carving (256B aligned)
    char* ws = (char*)d_ws;
    size_t off = 0;
    auto take = [&](size_t bytes) -> void* {
        void* p = ws + off;
        off += (bytes + 255) & ~(size_t)255;
        return p;
    };
    _Float16* xA     = (_Float16*)take((size_t)R * D * 2);
    _Float16* fc0wH  = (_Float16*)take((size_t)W * D * 2);
    _Float16* wih0H  = (_Float16*)take((size_t)G4 * W * 2);
    _Float16* whh0H  = (_Float16*)take((size_t)G4 * H * 2);
    _Float16* wih1H  = (_Float16*)take((size_t)G4 * H * 2);
    _Float16* whh1H  = (_Float16*)take((size_t)G4 * H * 2);
    _Float16* fcowH  = (_Float16*)take((size_t)OUT * H * 2);
    _Float16* wfno   = (_Float16*)take((size_t)W * W * 2);
    _Float16* hA     = (_Float16*)take((size_t)R * W * 2);
    _Float16* hB     = (_Float16*)take((size_t)R * W * 2);
    _Float16* seq0   = (_Float16*)take((size_t)R * H * 2);
    _Float16* Gx     = (_Float16*)take((size_t)R * G4 * 2);
    _Float16* hT1H   = (_Float16*)take((size_t)B * H * 2);
    _Float16* hx     = (_Float16*)take((size_t)3 * B * H * 2);
    float*    bsum0  = (float*)take((size_t)G4 * 4);
    float*    bsum1  = (float*)take((size_t)G4 * 4);
    unsigned* bars   = (unsigned*)take(64 * 4);

    auto cdiv = [](int a, int b) { return (a + b - 1) / b; };

    zero_u32<<<1, 64, 0, stream>>>(bars, 64);

    // f16 conversions of activations + weights
    cvt_f32_f16<<<cdiv(R * D, 256), 256, 0, stream>>>(x, xA, R * D);
    cvt_f32_f16<<<cdiv(W * D, 256), 256, 0, stream>>>(fc0_w, fc0wH, W * D);
    cvt_f32_f16<<<cdiv(G4 * W, 256), 256, 0, stream>>>(wih0, wih0H, G4 * W);
    cvt_f32_f16<<<cdiv(G4 * H, 256), 256, 0, stream>>>(whh0, whh0H, G4 * H);
    cvt_f32_f16<<<cdiv(G4 * H, 256), 256, 0, stream>>>(wih1, wih1H, G4 * H);
    cvt_f32_f16<<<cdiv(G4 * H, 256), 256, 0, stream>>>(whh1, whh1H, G4 * H);
    cvt_f32_f16<<<cdiv(OUT * H, 256), 256, 0, stream>>>(fcout_w, fcowH, OUT * H);
    bias_sum<<<cdiv(G4, 256), 256, 0, stream>>>(bih0, bhh0, bsum0, G4);
    bias_sum<<<cdiv(G4, 256), 256, 0, stream>>>(bih1, bhh1, bsum1, G4);

    // fc0: [R,64] @ [64->256] + bias -> hA (f16)
    {
        int blocks = cdiv((R / 16) * (W / 64), 8);
        wmma_gemm<0><<<blocks, 256, 0, stream>>>(xA, fc0wH, fc0_b, nullptr, hA, R, W, D);
    }
    // 4 fused FNO layers: h = gelu(h @ (Wr0 + lin^T) + lin_b)
    _Float16* cur = hA;
    _Float16* nxt = hB;
    for (int l = 0; l < 4; ++l) {
        fno_combine<<<cdiv(W * W, 256), 256, 0, stream>>>(spec_wr, lin_w, wfno, l);
        int blocks = cdiv((R / 16) * (W / 64), 8);
        wmma_gemm<1><<<blocks, 256, 0, stream>>>(cur, wfno, lin_b + l * W, nullptr, nxt,
                                                 R, W, W);
        _Float16* t = cur; cur = nxt; nxt = t;
    }

    // LSTM layer 0: batched input projection, then 8-WGP recurrence
    {
        int blocks = cdiv((R / 16) * (G4 / 64), 8);
        wmma_gemm<0><<<blocks, 256, 0, stream>>>(cur, wih0H, bsum0, nullptr, Gx, R, G4, W);
    }
    lstm_recur_mw<true><<<8, 1024, 0, stream>>>(Gx, whh0H, h0, c0, hx, bars + 0,
                                                seq0, h1, c1, nullptr, S);

    // LSTM layer 1
    {
        int blocks = cdiv((R / 16) * (G4 / 64), 8);
        wmma_gemm<0><<<blocks, 256, 0, stream>>>(seq0, wih1H, bsum1, nullptr, Gx, R, G4, H);
    }
    lstm_recur_mw<false><<<8, 1024, 0, stream>>>(Gx, whh1H, h0 + B * H, c0 + B * H, hx,
                                                 bars + 16, nullptr, h2, c2, hT1H, S);

    // fcout: [32,512] @ [512->64] + bias -> out (f32)
    {
        int blocks = cdiv((B / 16) * (OUT / 64), 8);   // = 1
        wmma_gemm<0><<<blocks, 256, 0, stream>>>(hT1H, fcowH, fcout_b, out_, nullptr,
                                                 B, OUT, H);
    }
}
